// MoEBlock_26560077758610
// MI455X (gfx1250) — matrix operand order repaired and checked by reading, not compiled, not run on hardware
//
#include <hip/hip_runtime.h>

// ---------------- problem constants ----------------
#define BB   4
#define TT   1024
#define CC   1024
#define NHH  16
#define HDD  64
#define EE   8
#define FFF  4096
#define MTOT (BB*TT)   // 4096 rows

// ---------------- WMMA types ----------------
typedef __attribute__((ext_vector_type(16))) __bf16 bf16x16;
typedef __attribute__((ext_vector_type(8)))  float  f32x8;
typedef __attribute__((ext_vector_type(4)))  unsigned int u32x4;
typedef __attribute__((ext_vector_type(8)))  unsigned int u32x8;

union AFrag { bf16x16 v; unsigned int u[8]; u32x4 q[2]; };
union FFrag { f32x8   v; float        f[8]; };

__device__ __forceinline__ f32x8 wmma_bf16(bf16x16 a, bf16x16 b, f32x8 c) {
  return __builtin_amdgcn_wmma_f32_16x16x32_bf16(
      /*neg_a=*/false, a, /*neg_b=*/false, b,
      /*c_mod=*/(short)0, c, /*reuse_a=*/false, /*reuse_b=*/false);
}

__device__ __forceinline__ unsigned short f2bf(float f) {
  unsigned int u = __float_as_uint(f);
  return (unsigned short)((u + 0x7FFFu + ((u >> 16) & 1u)) >> 16);  // RNE
}
__device__ __forceinline__ float bf2f(unsigned short s) {
  return __uint_as_float(((unsigned int)s) << 16);
}

// ---------------------------------------------------------------------
// TDM: 2D tile load global->LDS (bf16 elements, data_size=2B).
// tile_d0 elements per row, tile_d1 rows, row stride in elements.
// Pad: +4 DWORDs after every 16 DWORDs stored -> LDS row pitch 20 dwords
// (keeps rows 16B-aligned for b128 fragment loads, bank-conflict-free).
// Descriptor per CDNA5 ISA ch.8 (D# group0 + group1; groups 2/3 zero).
// ---------------------------------------------------------------------
__device__ __forceinline__ void tdm_load_2d_bf16(
    unsigned lds_off, unsigned long long gaddr,
    unsigned tile_d0, unsigned tile_d1, unsigned long long stride_elems) {
  u32x4 g0;
  g0[0] = 1u;                                   // count=1 (valid), user mode
  g0[1] = lds_off;                              // lds_addr
  g0[2] = (unsigned)gaddr;                      // global_addr[31:0]
  g0[3] = (unsigned)((gaddr >> 32) & 0x1FFFFFFull) | (2u << 30);  // addr[56:32] | type=2
  u32x8 g1;
  g1[0] = (1u << 16)                            // data_size=1 -> 2 bytes
        | (1u << 20)                            // pad_enable
        | (3u << 22)                            // pad_interval: every 16 DWORDs
        | (3u << 25);                           // pad_amount: 4 DWORDs
  g1[1] = (tile_d0 & 0xFFFFu) << 16;            // tensor_dim0[15:0] = tile_d0
  g1[2] = (tile_d0 >> 16) | ((tile_d1 & 0xFFFFu) << 16);   // tensor_dim1 = tile_d1
  g1[3] = (tile_d1 >> 16) | ((tile_d0 & 0xFFFFu) << 16);   // tile_dim0
  g1[4] = tile_d1 & 0xFFFFu;                    // tile_dim1 (tile_dim2=0)
  g1[5] = (unsigned)stride_elems;               // tensor_dim0_stride[31:0]
  g1[6] = (unsigned)((stride_elems >> 32) & 0xFFFFull);    // stride[47:32]
  g1[7] = 0;
  u32x4 gz = {0u, 0u, 0u, 0u};                  // groups 2/3: dims 2..4 unused
  asm volatile("tensor_load_to_lds %0, %1, %2, %3"
               :: "s"(g0), "s"(g1), "s"(gz), "s"(gz) : "memory");
}

// =====================================================================
// LayerNorm: one block per row of C=1024, 256 threads, bf16 output
// =====================================================================
__global__ __launch_bounds__(256)
void ln_kernel(const float* __restrict__ x, const float* __restrict__ g,
               const float* __restrict__ b, unsigned short* __restrict__ out) {
  int row = blockIdx.x;
  int t = threadIdx.x;
  const float* xr = x + (long)row * CC;
  float v[4], s1 = 0.f, s2 = 0.f;
#pragma unroll
  for (int i = 0; i < 4; i++) { v[i] = xr[t + i * 256]; s1 += v[i]; s2 += v[i] * v[i]; }
#pragma unroll
  for (int off = 16; off; off >>= 1) { s1 += __shfl_xor(s1, off, 32); s2 += __shfl_xor(s2, off, 32); }
  __shared__ float r1[8], r2[8];
  int wave = t >> 5, lane = t & 31;
  if (lane == 0) { r1[wave] = s1; r2[wave] = s2; }
  __syncthreads();
  if (t == 0) {
    float a = 0.f, c = 0.f;
#pragma unroll
    for (int i = 0; i < 8; i++) { a += r1[i]; c += r2[i]; }
    r1[0] = a; r2[0] = c;
  }
  __syncthreads();
  float mean = r1[0] * (1.f / CC);
  float var  = r2[0] * (1.f / CC) - mean * mean;
  float rstd = rsqrtf(var + 1e-5f);
  unsigned short* orow = out + (long)row * CC;
#pragma unroll
  for (int i = 0; i < 4; i++) {
    int c = t + i * 256;
    orow[c] = f2bf((v[i] - mean) * rstd * g[c] + b[c]);
  }
}

// =====================================================================
// Generic WMMA GEMM:  out = A[M,K](bf16) x W[N,K]^T(f32, cvt->bf16)
// Block tile 128x64, 8 waves, each wave 32x32 (2x2 WMMA frags), K-step 32.
// A: TDM DMA, triple-buffered (TENSORcnt). B: VALU f32->bf16, double-buffered.
// One barrier per K-step; fragments loaded as b128.
// EPI: 0 = bf16 out                       (QKV projections)
//      1 = f32 out + bias + residual      (Wo -> d_out)
//      2 = GELU(v+bias) -> bf16           (MoE fc)
//      3 = d_out += wsel[row,e]*(v+bias)  (MoE pj accumulate)
// =====================================================================
#define APITCH 20   // dwords per 32-element bf16 row (16 data + 4 TDM pad)

template<int EPI>
__global__ __launch_bounds__(256)
void gemm_kernel(const unsigned short* __restrict__ A, const float* __restrict__ W,
                 const float* __restrict__ bias, const float* __restrict__ resid,
                 const float* __restrict__ wsel, int expert,
                 float* __restrict__ outF, unsigned short* __restrict__ outB,
                 int M, int N, int K) {
  __shared__ __align__(16) unsigned int As[3][128 * APITCH];  // TDM triple buffer
  __shared__ __align__(16) unsigned int Bs[2][64 * APITCH];   // VALU double buffer
  int t = threadIdx.x;
  int lane = t & 31, wave = t >> 5;
  int n16 = lane & 15, half = lane >> 4;
  int wm = wave >> 1, wn = wave & 1;
  long bM = (long)blockIdx.y * 128;
  long bN = (long)blockIdx.x * 64;
  int nk = K >> 5;

  FFrag acc[2][2];
#pragma unroll
  for (int i = 0; i < 2; i++)
#pragma unroll
    for (int j = 0; j < 2; j++)
#pragma unroll
      for (int r = 0; r < 8; r++) acc[i][j].f[r] = 0.f;

  unsigned long long Abase = (unsigned long long)(uintptr_t)A + (unsigned long long)bM * K * 2ull;
  unsigned ldsA[3];
#pragma unroll
  for (int i = 0; i < 3; i++) ldsA[i] = (unsigned)(uintptr_t)&As[i][0];

  // prime: TDM tile 0 -> As[0]
  if (wave == 0) {
    tdm_load_2d_bf16(ldsA[0], Abase, 32u, 128u, (unsigned long long)K);
  }

  for (int kt = 0; kt < nk; kt++) {
    int cur = kt % 3;
    // issue TDM for next K-step (buffer (kt+1)%3: its last readers finished
    // before the barrier of iteration kt-1, so DMA may overlap freely)
    if (wave == 0 && (kt + 1) < nk) {
      tdm_load_2d_bf16(ldsA[(kt + 1) % 3],
                       Abase + (unsigned long long)(kt + 1) * 64ull,
                       32u, 128u, (unsigned long long)K);
    }
    // stage B (f32 -> packed bf16) into Bs[kt&1], prefetch next K-step of W
    unsigned int* Bw = &Bs[kt & 1][0];
#pragma unroll
    for (int i = 0; i < 4; i++) {
      int idx = t + i * 256;
      int r = idx >> 4, p = idx & 15;
      long wb = (bN + r) * (long)K + kt * 32 + 2 * p;
      Bw[r * APITCH + p] = (unsigned int)f2bf(W[wb]) | ((unsigned int)f2bf(W[wb + 1]) << 16);
      if (kt + 1 < nk) __builtin_prefetch(&W[wb + 32], 0, 0);
    }
    // current A tile must be resident before the barrier
    if (wave == 0) {
      if (kt + 1 < nk) __builtin_amdgcn_s_wait_tensorcnt(1);
      else             __builtin_amdgcn_s_wait_tensorcnt(0);
    }
    __syncthreads();

    const unsigned int* Ab = &As[cur][0];
    AFrag a[2], bfr[2];
#pragma unroll
    for (int mi = 0; mi < 2; mi++) {
      int m = wm * 32 + mi * 16 + n16;
      a[mi].q[0] = *(const u32x4*)&Ab[m * APITCH + 4 * half];       // pairs 4h..4h+3
      a[mi].q[1] = *(const u32x4*)&Ab[m * APITCH + 8 + 4 * half];   // pairs 8+4h..
    }
#pragma unroll
    for (int ni = 0; ni < 2; ni++) {
      int n = wn * 32 + ni * 16 + n16;
      bfr[ni].q[0] = *(const u32x4*)&Bw[n * APITCH + 4 * half];     // pairs 4h..4h+3
      bfr[ni].q[1] = *(const u32x4*)&Bw[n * APITCH + 8 + 4 * half];
    }
#pragma unroll
    for (int mi = 0; mi < 2; mi++)
#pragma unroll
      for (int ni = 0; ni < 2; ni++)
        acc[mi][ni].v = wmma_bf16(a[mi].v, bfr[ni].v, acc[mi][ni].v);
  }

  // epilogue (C layout: row = r + 8*half, col = n16)
#pragma unroll
  for (int mi = 0; mi < 2; mi++)
#pragma unroll
    for (int ni = 0; ni < 2; ni++) {
      long col = bN + wn * 32 + ni * 16 + n16;
      float bi = (EPI != 0 && bias) ? bias[col] : 0.f;
#pragma unroll
      for (int r = 0; r < 8; r++) {
        long row = bM + wm * 32 + mi * 16 + r + 8 * half;
        float vv = acc[mi][ni].f[r];
        if constexpr (EPI == 0) {
          outB[row * N + col] = f2bf(vv);
        } else if constexpr (EPI == 1) {
          outF[row * N + col] = vv + bi + resid[row * N + col];
        } else if constexpr (EPI == 2) {
          float gx = vv + bi;
          gx = 0.5f * gx * (1.0f + erff(gx * 0.70710678118f));
          outB[row * N + col] = f2bf(gx);
        } else {
          float wgt = wsel[row * EE + expert];
          if (wgt != 0.f) outF[row * N + col] += wgt * (vv + bi);
        }
      }
    }
}

// =====================================================================
// Flash attention: 1 wave / (batch, head, 16 q-rows). WMMA for QK^T & PV,
// online softmax with half-wave shfl reductions, P restaged via LDS.
// =====================================================================
__global__ __launch_bounds__(32)
void attn_kernel(const unsigned short* __restrict__ q,
                 const unsigned short* __restrict__ k,
                 const unsigned short* __restrict__ v,
                 unsigned short* __restrict__ y) {
  int gid = blockIdx.x;
  int qt = gid % (TT / 16);
  int tmp = gid / (TT / 16);
  int hh = tmp % NHH;
  int b  = tmp / NHH;
  int lane = threadIdx.x;
  int n16 = lane & 15, half = lane >> 4;
  int qrow0 = qt * 16;
  int headoff = hh * HDD;

  // Q A-fragments (two K-steps over HD=64), b128 loads
  AFrag qa[2];
  {
    const unsigned int* qu = (const unsigned int*)q;
    long qb = (((long)(b * TT + qrow0 + n16) * CC) + headoff) >> 1;
#pragma unroll
    for (int kk = 0; kk < 2; kk++) {
      qa[kk].q[0] = *(const u32x4*)&qu[qb + kk * 16 + 4 * half];
      qa[kk].q[1] = *(const u32x4*)&qu[qb + kk * 16 + 8 + 4 * half];
    }
  }

  float rm[8], rl[8];
  FFrag o[4];
#pragma unroll
  for (int r = 0; r < 8; r++) { rm[r] = -1e30f; rl[r] = 0.f; }
#pragma unroll
  for (int i = 0; i < 4; i++)
#pragma unroll
    for (int r = 0; r < 8; r++) o[i].f[r] = 0.f;

  __shared__ __align__(16) unsigned short Pl[16][2 * APITCH];  // pitch 20 dwords
  const unsigned int*  ku = (const unsigned int*)k;
  const unsigned short* vs = v;

  int ktmax = (qrow0 + 15) >> 5;
  for (int kt = 0; kt <= ktmax; kt++) {
    int key0 = kt * 32;
    FFrag s[2];
#pragma unroll
    for (int nt = 0; nt < 2; nt++) {
#pragma unroll
      for (int r = 0; r < 8; r++) s[nt].f[r] = 0.f;
      int key = key0 + nt * 16 + n16;
      long kb = (((long)(b * TT + key) * CC) + headoff) >> 1;
#pragma unroll
      for (int kk = 0; kk < 2; kk++) {
        AFrag bf;
        bf.q[0] = *(const u32x4*)&ku[kb + kk * 16 + 4 * half];
        bf.q[1] = *(const u32x4*)&ku[kb + kk * 16 + 8 + 4 * half];
        s[nt].v = wmma_bf16(qa[kk].v, bf.v, s[nt].v);
      }
    }
    // scale + causal mask + online softmax
#pragma unroll
    for (int r = 0; r < 8; r++) {
      int row = qrow0 + r + 8 * half;
      float mx = -1e30f;
#pragma unroll
      for (int nt = 0; nt < 2; nt++) {
        int key = key0 + nt * 16 + n16;
        float val = s[nt].f[r] * 0.125f;      // 1/sqrt(64)
        if (key > row) val = -1e30f;
        s[nt].f[r] = val;
        mx = fmaxf(mx, val);
      }
#pragma unroll
      for (int off = 1; off < 16; off <<= 1) mx = fmaxf(mx, __shfl_xor(mx, off, 32));
      float mnew = fmaxf(rm[r], mx);
      float alpha = __expf(rm[r] - mnew);
      rm[r] = mnew;
      float sum = 0.f;
#pragma unroll
      for (int nt = 0; nt < 2; nt++) {
        float p = __expf(s[nt].f[r] - mnew);
        s[nt].f[r] = p;
        sum += p;
      }
#pragma unroll
      for (int off = 1; off < 16; off <<= 1) sum += __shfl_xor(sum, off, 32);
      rl[r] = rl[r] * alpha + sum;
#pragma unroll
      for (int i = 0; i < 4; i++) o[i].f[r] *= alpha;
      // stash P (bf16) for layout conversion
      Pl[r + 8 * half][n16]      = f2bf(s[0].f[r]);
      Pl[r + 8 * half][16 + n16] = f2bf(s[1].f[r]);
    }
    __syncthreads();
    // reload P as A-fragment (16 x 32 over key dim), b128 loads
    AFrag pf;
    const unsigned int* pl = (const unsigned int*)&Pl[0][0];
    pf.q[0] = *(const u32x4*)&pl[n16 * APITCH + 8 * half];
    pf.q[1] = *(const u32x4*)&pl[n16 * APITCH + 8 * half + 4];
    // O += P x V over 4 hd-tiles
#pragma unroll
    for (int nt2 = 0; nt2 < 4; nt2++) {
      AFrag vf;
      int hd = headoff + nt2 * 16 + n16;
#pragma unroll
      for (int j = 0; j < 8; j++) {
        int klo = key0 + 2 * j + 16 * half;
        unsigned int lo = vs[(long)(b * TT + klo)     * CC + hd];
        unsigned int hi = vs[(long)(b * TT + klo + 1) * CC + hd];
        vf.u[j] = lo | (hi << 16);
      }
      o[nt2].v = wmma_bf16(pf.v, vf.v, o[nt2].v);
    }
    __syncthreads();
  }
  // finalize
#pragma unroll
  for (int r = 0; r < 8; r++) {
    float inv = 1.0f / rl[r];
    long row = (long)(b * TT + qrow0 + r + 8 * half);
#pragma unroll
    for (int i = 0; i < 4; i++)
      y[row * CC + headoff + i * 16 + n16] = f2bf(o[i].f[r] * inv);
  }
}

// =====================================================================
// Gating: 1 wave per token -> dense [M,8] top-2 softmax-weight matrix
// =====================================================================
__global__ __launch_bounds__(32)
void gate_kernel(const unsigned short* __restrict__ hb, const float* __restrict__ gW,
                 const float* __restrict__ gb, float* __restrict__ wsel) {
  int tok = blockIdx.x;
  int lane = threadIdx.x;
  float acc[EE];
#pragma unroll
  for (int e = 0; e < EE; e++) acc[e] = 0.f;
  const unsigned short* h = hb + (long)tok * CC;
  for (int c = lane; c < CC; c += 32) {
    float hv = bf2f(h[c]);
#pragma unroll
    for (int e = 0; e < EE; e++) acc[e] += hv * gW[e * CC + c];
  }
#pragma unroll
  for (int e = 0; e < EE; e++)
#pragma unroll
    for (int off = 16; off; off >>= 1) acc[e] += __shfl_xor(acc[e], off, 32);
  if (lane == 0) {
    float mx = -1e30f;
#pragma unroll
    for (int e = 0; e < EE; e++) { acc[e] += gb[e]; mx = fmaxf(mx, acc[e]); }
    float p[EE], sum = 0.f;
#pragma unroll
    for (int e = 0; e < EE; e++) { p[e] = __expf(acc[e] - mx); sum += p[e]; }
    float inv = 1.0f / sum;
    int i0 = 0;
#pragma unroll
    for (int e = 1; e < EE; e++) if (p[e] > p[i0]) i0 = e;
    int i1 = (i0 == 0) ? 1 : 0;
#pragma unroll
    for (int e = 0; e < EE; e++) if (e != i0 && p[e] > p[i1]) i1 = e;
#pragma unroll
    for (int e = 0; e < EE; e++) wsel[(long)tok * EE + e] = 0.f;
    wsel[(long)tok * EE + i0] = p[i0] * inv;
    wsel[(long)tok * EE + i1] = p[i1] * inv;
  }
}

// =====================================================================
extern "C" void kernel_launch(void* const* d_in, const int* in_sizes, int n_in,
                              void* d_out, int out_size, void* d_ws, size_t ws_size,
                              hipStream_t stream) {
  const float* x      = (const float*)d_in[0];
  const float* ln1_g  = (const float*)d_in[1];
  const float* ln1_b  = (const float*)d_in[2];
  const float* Wq     = (const float*)d_in[3];
  const float* Wk     = (const float*)d_in[4];
  const float* Wv     = (const float*)d_in[5];
  const float* Wo     = (const float*)d_in[6];
  const float* bo     = (const float*)d_in[7];
  const float* ln2_g  = (const float*)d_in[8];
  const float* ln2_b  = (const float*)d_in[9];
  const float* gate_W = (const float*)d_in[10];
  const float* gate_b = (const float*)d_in[11];
  const float* fc_W   = (const float*)d_in[12];
  const float* fc_b   = (const float*)d_in[13];
  const float* pj_W   = (const float*)d_in[14];
  const float* pj_b   = (const float*)d_in[15];
  float* out = (float*)d_out;

  // workspace carve-up (~72 MB)
  unsigned short* hb = (unsigned short*)d_ws;          // [M,C] bf16 (ln1, later ln2)
  unsigned short* qb = hb + (size_t)MTOT * CC;
  unsigned short* kb = qb + (size_t)MTOT * CC;
  unsigned short* vb = kb + (size_t)MTOT * CC;
  unsigned short* yb = vb + (size_t)MTOT * CC;
  unsigned short* h1 = yb + (size_t)MTOT * CC;         // [M,FF] bf16 per-expert temp
  float* wsel = (float*)(h1 + (size_t)MTOT * FFF);     // [M,E]

  // 1) LN1
  ln_kernel<<<MTOT, 256, 0, stream>>>(x, ln1_g, ln1_b, hb);

  // 2) Q,K,V projections (bf16 out)
  dim3 gQKV(CC / 64, MTOT / 128);
  gemm_kernel<0><<<gQKV, 256, 0, stream>>>(hb, Wq, nullptr, nullptr, nullptr, 0, nullptr, qb, MTOT, CC, CC);
  gemm_kernel<0><<<gQKV, 256, 0, stream>>>(hb, Wk, nullptr, nullptr, nullptr, 0, nullptr, kb, MTOT, CC, CC);
  gemm_kernel<0><<<gQKV, 256, 0, stream>>>(hb, Wv, nullptr, nullptr, nullptr, 0, nullptr, vb, MTOT, CC, CC);

  // 3) attention
  attn_kernel<<<BB * NHH * (TT / 16), 32, 0, stream>>>(qb, kb, vb, yb);

  // 4) output projection + bias + residual -> d_out (x1)
  gemm_kernel<1><<<gQKV, 256, 0, stream>>>(yb, Wo, bo, x, nullptr, 0, out, nullptr, MTOT, CC, CC);

  // 5) LN2 (reads x1 from d_out), gating
  ln_kernel<<<MTOT, 256, 0, stream>>>(out, ln2_g, ln2_b, hb);
  gate_kernel<<<MTOT, 32, 0, stream>>>(hb, gate_W, gate_b, wsel);

  // 6) MoE experts: fc+GELU -> h1 (bf16), then pj accumulate into d_out
  dim3 gFC(FFF / 64, MTOT / 128);
  dim3 gPJ(CC / 64, MTOT / 128);
  for (int e = 0; e < EE; e++) {
    gemm_kernel<2><<<gFC, 256, 0, stream>>>(hb, fc_W + (size_t)e * FFF * CC,
                                            fc_b + (size_t)e * FFF, nullptr, nullptr, 0,
                                            nullptr, h1, MTOT, FFF, CC);
    gemm_kernel<3><<<gPJ, 256, 0, stream>>>(h1, pj_W + (size_t)e * CC * FFF,
                                            pj_b + (size_t)e * CC, nullptr, wsel, e,
                                            out, nullptr, MTOT, CC, FFF);
  }
}
